// Novel_Node_GAT_N_Sim_51101520888524
// MI455X (gfx1250) — compile-verified, matrix operand order checked by reading, not run on hardware
//
#include <hip/hip_runtime.h>
#include <hip/hip_bf16.h>

#define N_NODES 50000
#define N_EDGES 800000
#define IN_FEAT 128
#define HID 64
#define CLS 40
#define NEG_BIG -3.402823466e38f

typedef __attribute__((ext_vector_type(16))) __bf16 v16bf;
typedef __attribute__((ext_vector_type(8)))  __bf16 v8bf;
typedef __attribute__((ext_vector_type(8)))  float  v8f;

__device__ __forceinline__ __bf16 f2bf(float f) {
    union { float f; unsigned u; } a; a.f = f;
    unsigned r = a.u + 0x7FFFu + ((a.u >> 16) & 1u);   // round-to-nearest-even
    unsigned short h = (unsigned short)(r >> 16);
    __bf16 b;
    __builtin_memcpy(&b, &h, sizeof(h));
    return b;
}

// Coalesced one-shot f32 -> bf16 conversion (count % 4 == 0)
__global__ void cvt_f32_bf16(const float* __restrict__ in, __bf16* __restrict__ out,
                             long count) {
    long i = ((long)blockIdx.x * blockDim.x + threadIdx.x) * 4;
    if (i >= count) return;
    float4 v = *(const float4*)(in + i);
    out[i]     = f2bf(v.x);
    out[i + 1] = f2bf(v.y);
    out[i + 2] = f2bf(v.z);
    out[i + 3] = f2bf(v.w);
}

// Build a 16x32 bf16 A/B fragment: two contiguous 16B chunks per lane
// (K groups 0..7 and 16..23 for half=0; 8..15 and 24..31 for half=1)
__device__ __forceinline__ v16bf ld_frag(const __bf16* __restrict__ p) {
    v8bf lo = *(const v8bf*)(p);
    v8bf hi = *(const v8bf*)(p + 16);
    return __builtin_shufflevector(lo, hi, 0, 1, 2, 3, 4, 5, 6, 7,
                                           8, 9, 10, 11, 12, 13, 14, 15);
}

// H[N, HID] = A[N, K] @ W[HID, K]^T  (bf16 WMMA, f32 accumulate)
// grid.x = N/16 row tiles; block = 128 threads = 4 waves; wave w -> col tile w.
// W staged into LDS with async-to-LDS DMA; A streamed from global (read once).
template <int K>
__global__ void gemm_xWT_wmma(const __bf16* __restrict__ A,
                              const __bf16* __restrict__ Wb,
                              float* __restrict__ H) {
    __shared__ __bf16 Wlds[HID * K];

    // cooperative async stage of W (HID*K bf16) into LDS, 16B per lane per op
    constexpr int NCHUNK = HID * K / 8;          // 16B chunks; multiple of 128
    for (int c = threadIdx.x; c < NCHUNK; c += 128) {
        unsigned ldsoff = (unsigned)(unsigned long long)(Wlds + c * 8);
        unsigned long long ga = (unsigned long long)(Wb + c * 8);
        asm volatile("global_load_async_to_lds_b128 %0, %1, off"
                     :: "v"(ldsoff), "v"(ga) : "memory");
    }
    asm volatile("s_wait_asynccnt 0" ::: "memory");
    __syncthreads();

    const int lane = threadIdx.x & 31;
    const int wave = threadIdx.x >> 5;
    const int m0   = blockIdx.x * 16;
    const int n0   = wave * 16;
    const int half = lane >> 4;                  // 0 or 1
    const __bf16* arow = A + (long)(m0 + (lane & 15)) * K + half * 8;
    const __bf16* brow = Wlds + (n0 + (lane & 15)) * K + half * 8;

    v8f c = {};
#pragma unroll
    for (int kb = 0; kb < K; kb += 32) {
        v16bf a = ld_frag(arow + kb);            // global_load_b128 x2
        v16bf b = ld_frag(brow + kb);            // ds_load_b128 x2
        c = __builtin_amdgcn_wmma_f32_16x16x32_bf16(false, a, false, b,
                                                    (short)0, c, false, false);
    }
#pragma unroll
    for (int r = 0; r < 8; ++r) {                // C/D: VGPR r, lanes 16-31 -> M+8
        int row = m0 + r + half * 8;
        H[(long)row * HID + n0 + (lane & 15)] = c[r];
    }
}

__global__ void deg_init(float* __restrict__ outdeg, float* __restrict__ indeg) {
    int n = blockIdx.x * blockDim.x + threadIdx.x;
    if (n < N_NODES) { outdeg[n] = 1.0f; indeg[n] = 1.0f; }   // self-loop
}

__global__ void deg_count(const int* __restrict__ ei,
                          float* __restrict__ outdeg, float* __restrict__ indeg) {
    int e = blockIdx.x * blockDim.x + threadIdx.x;
    if (e >= N_EDGES) return;
    atomicAdd(&outdeg[ei[e]], 1.0f);
    atomicAdd(&indeg[ei[N_EDGES + e]], 1.0f);
}

__global__ void layer_init(float* __restrict__ agg,
                           float* __restrict__ msim, float* __restrict__ mnsim,
                           float* __restrict__ ssim, float* __restrict__ snsim) {
    int i = blockIdx.x * blockDim.x + threadIdx.x;
    if (i < N_NODES * HID) agg[i] = NEG_BIG;
    if (i < N_NODES) {
        msim[i] = NEG_BIG; mnsim[i] = NEG_BIG;
        ssim[i] = 0.0f;    snsim[i] = 0.0f;
    }
}

// per-node scalar projections: ps=<aw[0:H],h>, pt=<aw[H:2H],h>, qs/qt for sw
__global__ void node_scores(const float* __restrict__ h,
                            const float* __restrict__ aw, const float* __restrict__ sw,
                            float* __restrict__ ps, float* __restrict__ pt,
                            float* __restrict__ qs, float* __restrict__ qt) {
    int n = blockIdx.x * blockDim.x + threadIdx.x;
    if (n >= N_NODES) return;
    const float* hp = h + (long)n * HID;
    float a0 = 0.f, a1 = 0.f, b0 = 0.f, b1 = 0.f;
#pragma unroll 8
    for (int k = 0; k < HID; ++k) {
        float v = hp[k];
        a0 += v * aw[k];       a1 += v * aw[HID + k];
        b0 += v * sw[k];       b1 += v * sw[HID + k];
    }
    ps[n] = a0; pt[n] = a1; qs[n] = b0; qt[n] = b1;
}

__device__ __forceinline__ void edge_ids(int e, const int* __restrict__ ei,
                                         int& src, int& dst) {
    if (e < N_EDGES) { src = ei[e]; dst = ei[N_EDGES + e]; }
    else             { src = e - N_EDGES; dst = e - N_EDGES; }   // self loop
}

// leaky-relu logits per edge + segment max over src
__global__ void edge_logits(const int* __restrict__ ei, const float* __restrict__ norm,
                            const float* __restrict__ ps, const float* __restrict__ pt,
                            const float* __restrict__ qs, const float* __restrict__ qt,
                            const float* __restrict__ outdeg, const float* __restrict__ indeg,
                            const float* __restrict__ ab, const float* __restrict__ sb,
                            float* __restrict__ sim, float* __restrict__ nsim,
                            float* __restrict__ msim, float* __restrict__ mnsim) {
    int e = blockIdx.x * blockDim.x + threadIdx.x;
    if (e >= N_EDGES + N_NODES) return;
    int src, dst; edge_ids(e, ei, src, dst);
    float s = norm[e] * (ps[src] + pt[dst]) + ab[0];
    s = s > 0.f ? s : 0.2f * s;
    float t = outdeg[src] * qs[src] + indeg[dst] * qt[dst] + sb[0];
    t = t > 0.f ? t : 0.2f * t;
    sim[e] = s; nsim[e] = t;
    atomicMax(&msim[src], s);        // global_atomic_max_num_f32
    atomicMax(&mnsim[src], t);
}

// exp(logit - segmax) in place + segment sums over src
__global__ void edge_exp(const int* __restrict__ ei,
                         float* __restrict__ sim, float* __restrict__ nsim,
                         const float* __restrict__ msim, const float* __restrict__ mnsim,
                         float* __restrict__ ssim, float* __restrict__ snsim) {
    int e = blockIdx.x * blockDim.x + threadIdx.x;
    if (e >= N_EDGES + N_NODES) return;
    int src, dst; edge_ids(e, ei, src, dst);
    (void)dst;
    float es = __expf(sim[e] - msim[src]);
    float en = __expf(nsim[e] - mnsim[src]);
    sim[e] = es; nsim[e] = en;
    atomicAdd(&ssim[src], es);
    atomicAdd(&snsim[src], en);
}

// wave-per-edge: w = alpha*a + (1-alpha)*struct;  agg[dst] = max(agg[dst], w*h[src])
__global__ void edge_aggregate(const float* __restrict__ h, const int* __restrict__ ei,
                               const float* __restrict__ esim, const float* __restrict__ ssim,
                               const float* __restrict__ ensim, const float* __restrict__ snsim,
                               const float* __restrict__ alpha_p,
                               float* __restrict__ agg) {
    int e = blockIdx.x * 8 + (threadIdx.x >> 5);
    if (e >= N_EDGES + N_NODES) return;
    int lane = threadIdx.x & 31;
    int src, dst; edge_ids(e, ei, src, dst);
    float alpha = alpha_p[0];
    float w = alpha * (esim[e] / ssim[src]) +
              (1.0f - alpha) * (ensim[e] / snsim[src]);
    const float* hp = h + (long)src * HID;
    float* op = agg + (long)dst * HID;
    __builtin_prefetch(hp, 0, 3);                  // global_prefetch_b8
    atomicMax(op + lane,      w * hp[lane]);       // segment_max scatter
    atomicMax(op + lane + 32, w * hp[lane + 32]);
}

// add bias, relu; write f32 (for classifier) + bf16 (next GEMM A)
__global__ void finalize_relu(const float* __restrict__ agg, const float* __restrict__ bias,
                              float* __restrict__ hout, __bf16* __restrict__ houtb) {
    int i = blockIdx.x * blockDim.x + threadIdx.x;
    if (i >= N_NODES * HID) return;
    float v = agg[i] + bias[i & (HID - 1)];
    v = v > 0.f ? v : 0.f;
    hout[i] = v;
    houtb[i] = f2bf(v);
}

// logits = h @ out_w^T + b; out = log_softmax(logits)
__global__ void classify(const float* __restrict__ h, const float* __restrict__ Wc,
                         const float* __restrict__ bc, float* __restrict__ out) {
    int n = blockIdx.x * blockDim.x + threadIdx.x;
    if (n >= N_NODES) return;
    const float4* hp = (const float4*)(h + (long)n * HID);
    float logits[CLS];
    float mx = NEG_BIG;
#pragma unroll 4
    for (int c = 0; c < CLS; ++c) {
        const float4* wp = (const float4*)(Wc + (long)c * HID);
        float acc = bc[c];
#pragma unroll
        for (int i = 0; i < HID / 4; ++i) {
            float4 hv = hp[i], wv = wp[i];
            acc += hv.x * wv.x + hv.y * wv.y + hv.z * wv.z + hv.w * wv.w;
        }
        logits[c] = acc;
        mx = fmaxf(mx, acc);
    }
    float s = 0.f;
#pragma unroll
    for (int c = 0; c < CLS; ++c) s += __expf(logits[c] - mx);
    float ls = __logf(s) + mx;
    float* op = out + (long)n * CLS;
#pragma unroll
    for (int c = 0; c < CLS; ++c) op[c] = logits[c] - ls;
}

extern "C" void kernel_launch(void* const* d_in, const int* in_sizes, int n_in,
                              void* d_out, int out_size, void* d_ws, size_t ws_size,
                              hipStream_t stream) {
    const float* x   = (const float*)d_in[0];
    const int*   ei  = (const int*)  d_in[1];
    const float* nrm = (const float*)d_in[2];
    const float* lw0 = (const float*)d_in[3];
    const float* lw1 = (const float*)d_in[4];
    const float* aw0 = (const float*)d_in[5];
    const float* ab0 = (const float*)d_in[6];
    const float* aw1 = (const float*)d_in[7];
    const float* ab1 = (const float*)d_in[8];
    const float* sw0 = (const float*)d_in[9];
    const float* sb0 = (const float*)d_in[10];
    const float* sw1 = (const float*)d_in[11];
    const float* sb1 = (const float*)d_in[12];
    const float* al0 = (const float*)d_in[13];
    const float* al1 = (const float*)d_in[14];
    const float* b0  = (const float*)d_in[15];
    const float* b1  = (const float*)d_in[16];
    const float* ow  = (const float*)d_in[17];
    const float* ob  = (const float*)d_in[18];
    float* out = (float*)d_out;

    const int Etot = N_EDGES + N_NODES;
    float* ws = (float*)d_ws;
    size_t off = 0;
    float* H      = ws + off; off += (size_t)N_NODES * HID;   // pre-activation h
    float* HACT   = ws + off; off += (size_t)N_NODES * HID;   // relu(layer out) f32
    float* AGG    = ws + off; off += (size_t)N_NODES * HID;   // segment-max accum
    float* outdeg = ws + off; off += N_NODES;
    float* indeg  = ws + off; off += N_NODES;
    float* ps     = ws + off; off += N_NODES;
    float* pt     = ws + off; off += N_NODES;
    float* qs     = ws + off; off += N_NODES;
    float* qt     = ws + off; off += N_NODES;
    float* msim   = ws + off; off += N_NODES;
    float* mnsim  = ws + off; off += N_NODES;
    float* ssim   = ws + off; off += N_NODES;
    float* snsim  = ws + off; off += N_NODES;
    float* sim    = ws + off; off += (size_t)Etot;
    float* nsim   = ws + off; off += (size_t)Etot;
    // bf16 region (16B-aligned by construction)
    __bf16* wsb  = (__bf16*)(ws + off);
    __bf16* XB    = wsb; wsb += (size_t)N_NODES * IN_FEAT;    // bf16(x)
    __bf16* HACTB = wsb; wsb += (size_t)N_NODES * HID;        // bf16(relu(layer1))
    __bf16* W0B   = wsb; wsb += (size_t)HID * IN_FEAT;
    __bf16* W1B   = wsb; wsb += (size_t)HID * HID;

    const int TB = 256;
    // one-shot bf16 conversions (coalesced)
    cvt_f32_bf16<<<((size_t)N_NODES * IN_FEAT / 4 + TB - 1) / TB, TB, 0, stream>>>(x, XB, (long)N_NODES * IN_FEAT);
    cvt_f32_bf16<<<(HID * IN_FEAT / 4 + TB - 1) / TB, TB, 0, stream>>>(lw0, W0B, HID * IN_FEAT);
    cvt_f32_bf16<<<(HID * HID / 4 + TB - 1) / TB, TB, 0, stream>>>(lw1, W1B, HID * HID);

    // degrees (graph-only; shared by both layers)
    deg_init <<<(N_NODES + TB - 1) / TB, TB, 0, stream>>>(outdeg, indeg);
    deg_count<<<(N_EDGES + TB - 1) / TB, TB, 0, stream>>>(ei, outdeg, indeg);

    auto run_edges = [&](const float* aw, const float* ab,
                         const float* sw, const float* sb,
                         const float* al, const float* bias) {
        layer_init    <<<(N_NODES * HID + TB - 1) / TB, TB, 0, stream>>>(AGG, msim, mnsim, ssim, snsim);
        node_scores   <<<(N_NODES + TB - 1) / TB, TB, 0, stream>>>(H, aw, sw, ps, pt, qs, qt);
        edge_logits   <<<(Etot + TB - 1) / TB, TB, 0, stream>>>(ei, nrm, ps, pt, qs, qt,
                                                               outdeg, indeg, ab, sb,
                                                               sim, nsim, msim, mnsim);
        edge_exp      <<<(Etot + TB - 1) / TB, TB, 0, stream>>>(ei, sim, nsim, msim, mnsim, ssim, snsim);
        edge_aggregate<<<(Etot + 7) / 8, TB, 0, stream>>>(H, ei, sim, ssim, nsim, snsim, al, AGG);
        finalize_relu <<<(N_NODES * HID + TB - 1) / TB, TB, 0, stream>>>(AGG, bias, HACT, HACTB);
    };

    // layer 1
    gemm_xWT_wmma<IN_FEAT><<<N_NODES / 16, 128, 0, stream>>>(XB, W0B, H);
    run_edges(aw0, ab0, sw0, sb0, al0, b0);
    // layer 2
    gemm_xWT_wmma<HID><<<N_NODES / 16, 128, 0, stream>>>(HACTB, W1B, H);
    run_edges(aw1, ab1, sw1, sb1, al1, b1);

    classify<<<(N_NODES + TB - 1) / TB, TB, 0, stream>>>(HACT, ow, ob, out);
}
